// RWKV_GPT_Block_44676249813131
// MI455X (gfx1250) — matrix operand order repaired and checked by reading, not compiled, not run on hardware
//
#include <hip/hip_runtime.h>
#include <hip/hip_bf16.h>

// ---------------- problem constants (match reference) ----------------
#define BB 4
#define TT 2048
#define HH 1024
#define HU 4096
#define MM (BB * TT)          // 8192 tokens
#define K_CLAMP 60.0f
#define K_EPS   1e-8f
#define LN_EPS  1e-5f

typedef __bf16 bf16;
typedef __attribute__((ext_vector_type(16))) __bf16 v16bf;
typedef __attribute__((ext_vector_type(8)))  float  v8f;

// ================= fp32 -> bf16 transpose (W[K,N] -> Wt[N,K]) =================
__global__ __launch_bounds__(256)
void transpose_cast(const float* __restrict__ src, bf16* __restrict__ dst,
                    int K, int N) {
    __shared__ float tile[32][33];
    int kb = blockIdx.x * 32, nb = blockIdx.y * 32;
    int tx = threadIdx.x, ty = threadIdx.y;   // 32 x 8
#pragma unroll
    for (int i = 0; i < 4; ++i)
        tile[ty + i * 8][tx] = src[(size_t)(kb + ty + i * 8) * N + nb + tx];
    __syncthreads();
#pragma unroll
    for (int i = 0; i < 4; ++i)
        dst[(size_t)(nb + ty + i * 8) * K + kb + tx] = (bf16)tile[tx][ty + i * 8];
}

// ================= LayerNorm over H=1024 (one block per token) ================
__global__ __launch_bounds__(256)
void ln_kernel(const float* __restrict__ src, float* __restrict__ dst,
               const float* __restrict__ g, const float* __restrict__ b) {
    const float* x = src + (size_t)blockIdx.x * HH;
    float*       y = dst + (size_t)blockIdx.x * HH;
    float vals[4], lsum = 0.f, lsq = 0.f;
#pragma unroll
    for (int i = 0; i < 4; ++i) {
        float v = x[threadIdx.x + i * 256];
        vals[i] = v; lsum += v; lsq += v * v;
    }
    __shared__ float ssum[256], ssq[256];
    ssum[threadIdx.x] = lsum; ssq[threadIdx.x] = lsq;
    __syncthreads();
    for (int s = 128; s > 0; s >>= 1) {
        if (threadIdx.x < s) {
            ssum[threadIdx.x] += ssum[threadIdx.x + s];
            ssq[threadIdx.x]  += ssq[threadIdx.x + s];
        }
        __syncthreads();
    }
    float mean = ssum[0] * (1.f / HH);
    float var  = ssq[0] * (1.f / HH) - mean * mean;
    float inv  = rsqrtf(var + LN_EPS);
#pragma unroll
    for (int i = 0; i < 4; ++i) {
        int h = threadIdx.x + i * 256;
        y[h] = (vals[i] - mean) * inv * g[h] + b[h];
    }
}

// ============ token-shift + lerp mixes -> bf16 GEMM activations ==============
__global__ __launch_bounds__(256)
void shift_mix3(const float* __restrict__ xn,
                const float* __restrict__ mk, const float* __restrict__ mv,
                const float* __restrict__ mr,
                bf16* __restrict__ xk, bf16* __restrict__ xv, bf16* __restrict__ xr) {
    size_t i = (size_t)blockIdx.x * 256 + threadIdx.x;   // over M*H
    int h = (int)(i % HH);
    int t = (int)((i / HH) % TT);
    float cur  = xn[i];
    float prev = (t > 0) ? xn[i - HH] : 0.f;
    float a;
    a = mk[h]; xk[i] = (bf16)(cur * a + prev * (1.f - a));
    a = mv[h]; xv[i] = (bf16)(cur * a + prev * (1.f - a));
    a = mr[h]; xr[i] = (bf16)(cur * a + prev * (1.f - a));
}

__global__ __launch_bounds__(256)
void shift_mix2(const float* __restrict__ xn,
                const float* __restrict__ mk, const float* __restrict__ mr,
                bf16* __restrict__ xk, bf16* __restrict__ xr) {
    size_t i = (size_t)blockIdx.x * 256 + threadIdx.x;
    int h = (int)(i % HH);
    int t = (int)((i / HH) % TT);
    float cur  = xn[i];
    float prev = (t > 0) ? xn[i - HH] : 0.f;
    float a;
    a = mk[h]; xk[i] = (bf16)(cur * a + prev * (1.f - a));
    a = mr[h]; xr[i] = (bf16)(cur * a + prev * (1.f - a));
}

// ====================== WKV linear recurrence over time ======================
// one thread per (batch, channel): 16384 channels = 512 wave32 -> latency hidden
__global__ __launch_bounds__(256)
void wkv_scan(const float* __restrict__ kf, const float* __restrict__ kvf,
              const bf16* __restrict__ srb,
              const float* __restrict__ decay_p, const float* __restrict__ first_p,
              bf16* __restrict__ rwkv) {
    int idx = blockIdx.x * 256 + threadIdx.x;   // B*HU
    int c = idx % HU, b = idx / HU;
    float decay = __expf(-__expf(decay_p[c]));
    float first = __expf(first_p[c]);
    float sk = 0.f, skv = 0.f;
    size_t base = (size_t)b * TT * HU + c;
    for (int t = 0; t < TT; ++t) {
        size_t off = base + (size_t)t * HU;
        float kt  = kf[off];
        float kvt = kvf[off];
        float wk  = first * kt  + sk;
        float wkv = first * kvt + skv;
        sk  = kt  + decay * sk;
        skv = kvt + decay * skv;
        float sr = (float)srb[off];
        rwkv[off] = (bf16)(sr * (wkv / (wk + K_EPS)));
    }
}

// ====================== bf16 WMMA GEMM with fused epilogues ==================
enum { EPI_EXPCLAMP = 0, EPI_MULBUF, EPI_SIGB, EPI_SILUB, EPI_RESID, EPI_RESID2, EPI_SIGF };

__device__ inline v16bf load_fragA(const bf16* p) {
    // A 16x32 bf16 layout: lane holds 16 consecutive K values (contiguous 32B)
    union { uint4 u[2]; v16bf v; } f;
    f.u[0] = *(const uint4*)(p);
    f.u[1] = *(const uint4*)(p + 8);
    return f.v;
}
__device__ inline v16bf load_fragB(const bf16* p) {
    // B 32x16 bf16 layout: lane holds 16 consecutive K values (contiguous 32B)
    union { uint4 u[2]; v16bf v; } f;
    f.u[0] = *(const uint4*)(p);
    f.u[1] = *(const uint4*)(p + 8);
    return f.v;
}

template <int EPI, typename OutT>
__global__ __launch_bounds__(128)
void gemm_bf16_wmma(const bf16* __restrict__ A,      // [M,K] row-major bf16
                    const bf16* __restrict__ Wt,     // [N,K] row-major bf16 (W^T)
                    OutT* __restrict__ C,            // [M,N]
                    const float* __restrict__ base1,
                    const float* __restrict__ base2,
                    int M, int N, int K) {
    const int lane = threadIdx.x & 31;
    const int wave = threadIdx.x >> 5;
    const int m0 = blockIdx.y * 128 + (wave >> 1) * 64;
    const int n0 = blockIdx.x * 128 + (wave & 1) * 64;
    const int lr  = lane & 15;
    const int khA = (lane >> 4) * 16;   // A lane-half K offset
    const int khB = (lane >> 4) * 16;   // B lane-half K offset

    v8f acc[4][4] = {};

    for (int k0 = 0; k0 < K; k0 += 32) {
        v16bf af[4], bf_[4];
#pragma unroll
        for (int mi = 0; mi < 4; ++mi)
            af[mi] = load_fragA(A + (size_t)(m0 + mi * 16 + lr) * K + k0 + khA);
#pragma unroll
        for (int ni = 0; ni < 4; ++ni)
            bf_[ni] = load_fragB(Wt + (size_t)(n0 + ni * 16 + lr) * K + k0 + khB);
#pragma unroll
        for (int mi = 0; mi < 4; ++mi)
#pragma unroll
            for (int ni = 0; ni < 4; ++ni)
                acc[mi][ni] = __builtin_amdgcn_wmma_f32_16x16x32_bf16(
                    false, af[mi], false, bf_[ni], (short)0, acc[mi][ni], false, false);
    }

    const int rbase = (lane >> 4) * 8;   // C: lanes16-31 hold M+8
#pragma unroll
    for (int mi = 0; mi < 4; ++mi)
#pragma unroll
        for (int ni = 0; ni < 4; ++ni)
#pragma unroll
            for (int e = 0; e < 8; ++e) {
                int row = m0 + mi * 16 + rbase + e;
                int col = n0 + ni * 16 + lr;
                size_t idx = (size_t)row * N + col;
                float v = acc[mi][ni][e];
                float o;
                if (EPI == EPI_EXPCLAMP)      o = __expf(fminf(v, K_CLAMP));
                else if (EPI == EPI_MULBUF)   o = v * base1[idx];
                else if (EPI == EPI_SIGB)     o = 1.f / (1.f + __expf(-v));
                else if (EPI == EPI_SILUB)    o = v / (1.f + __expf(-v));
                else if (EPI == EPI_RESID)    o = base1[idx] + v;
                else if (EPI == EPI_RESID2)   o = base1[idx] + base2[idx] + v;
                else /* EPI_SIGF */           o = 1.f / (1.f + __expf(-v));
                C[idx] = (OutT)o;
            }
}

// =============================== launcher ====================================
extern "C" void kernel_launch(void* const* d_in, const int* in_sizes, int n_in,
                              void* d_out, int out_size, void* d_ws, size_t ws_size,
                              hipStream_t stream) {
    const float* x       = (const float*)d_in[0];
    const float* ln0_g   = (const float*)d_in[1];
    const float* ln0_b   = (const float*)d_in[2];
    const float* pre_g   = (const float*)d_in[3];
    const float* pre_b   = (const float*)d_in[4];
    const float* post_g  = (const float*)d_in[5];
    const float* post_b  = (const float*)d_in[6];
    const float* tm_dec  = (const float*)d_in[7];
    const float* tm_fst  = (const float*)d_in[8];
    const float* tm_mk   = (const float*)d_in[9];
    const float* tm_mv   = (const float*)d_in[10];
    const float* tm_mr   = (const float*)d_in[11];
    const float* tm_Wk   = (const float*)d_in[12];
    const float* tm_Wv   = (const float*)d_in[13];
    const float* tm_Wr   = (const float*)d_in[14];
    const float* tm_Wo   = (const float*)d_in[15];
    const float* cm_mk   = (const float*)d_in[16];
    const float* cm_mr   = (const float*)d_in[17];
    const float* cm_Wk   = (const float*)d_in[18];
    const float* cm_Wv   = (const float*)d_in[19];
    const float* cm_Wr   = (const float*)d_in[20];
    float* out = (float*)d_out;

    // -------- workspace layout (~580 MB needed) --------
    size_t off = 0;
    auto alloc = [&](size_t bytes) { size_t o = off; off += (bytes + 255) & ~(size_t)255; return o; };
    const size_t MH_f  = (size_t)MM * HH * 4;
    const size_t MH_b  = (size_t)MM * HH * 2;
    const size_t MHU_f = (size_t)MM * HU * 4;
    const size_t MHU_b = (size_t)MM * HU * 2;
    const size_t W_hhu = (size_t)HH * HU * 2;

    char* ws = (char*)d_ws;
    bf16* tmWkT = (bf16*)(ws + alloc(W_hhu));           // [HU,HH]
    bf16* tmWvT = (bf16*)(ws + alloc(W_hhu));
    bf16* tmWrT = (bf16*)(ws + alloc(W_hhu));
    bf16* tmWoT = (bf16*)(ws + alloc(W_hhu));           // [HH,HU]
    bf16* cmWkT = (bf16*)(ws + alloc(W_hhu));           // [HU,HH]
    bf16* cmWvT = (bf16*)(ws + alloc(W_hhu));           // [HH,HU]
    bf16* cmWrT = (bf16*)(ws + alloc((size_t)HH * HH * 2)); // [HH,HH]

    float* xln0 = (float*)(ws + alloc(MH_f));
    float* xn   = (float*)(ws + alloc(MH_f));           // reused as xn2
    bf16* xkb   = (bf16*)(ws + alloc(MH_b));            // reused as xk2
    bf16* xvb   = (bf16*)(ws + alloc(MH_b));
    bf16* xrb   = (bf16*)(ws + alloc(MH_b));            // reused as xr2
    float* kf   = (float*)(ws + alloc(MHU_f));
    float* kvf  = (float*)(ws + alloc(MHU_f));
    bf16* srb   = (bf16*)(ws + alloc(MHU_b));
    bf16* rwkvb = (bf16*)(ws + alloc(MHU_b));
    float* x2f  = (float*)(ws + alloc(MH_f));
    bf16* kkb   = (bf16*)kf;                            // alias: k dead after scan
    float* rrf  = (float*)kvf;                          // alias: kv dead after scan

    dim3 tb(32, 8);
    // weight down-convert + transpose (bf16 for WMMA, K-contiguous rows)
    transpose_cast<<<dim3(HH/32, HU/32), tb, 0, stream>>>(tm_Wk, tmWkT, HH, HU);
    transpose_cast<<<dim3(HH/32, HU/32), tb, 0, stream>>>(tm_Wv, tmWvT, HH, HU);
    transpose_cast<<<dim3(HH/32, HU/32), tb, 0, stream>>>(tm_Wr, tmWrT, HH, HU);
    transpose_cast<<<dim3(HU/32, HH/32), tb, 0, stream>>>(tm_Wo, tmWoT, HU, HH);
    transpose_cast<<<dim3(HH/32, HU/32), tb, 0, stream>>>(cm_Wk, cmWkT, HH, HU);
    transpose_cast<<<dim3(HU/32, HH/32), tb, 0, stream>>>(cm_Wv, cmWvT, HU, HH);
    transpose_cast<<<dim3(HH/32, HH/32), tb, 0, stream>>>(cm_Wr, cmWrT, HH, HH);

    // ln0 + pre-LN + shift/mix
    ln_kernel<<<MM, 256, 0, stream>>>(x,    xln0, ln0_g, ln0_b);
    ln_kernel<<<MM, 256, 0, stream>>>(xln0, xn,   pre_g, pre_b);
    shift_mix3<<<(MM * HH) / 256, 256, 0, stream>>>(xn, tm_mk, tm_mv, tm_mr, xkb, xvb, xrb);

    // time-mix GEMMs (M=8192): k = exp(min(.,60)); kv = v*k fused; sig(r) -> bf16
    dim3 gHU(HU / 128, MM / 128), gH(HH / 128, MM / 128);
    gemm_bf16_wmma<EPI_EXPCLAMP, float><<<gHU, 128, 0, stream>>>(xkb, tmWkT, kf,  nullptr, nullptr, MM, HU, HH);
    gemm_bf16_wmma<EPI_MULBUF,   float><<<gHU, 128, 0, stream>>>(xvb, tmWvT, kvf, kf,      nullptr, MM, HU, HH);
    gemm_bf16_wmma<EPI_SIGB,     bf16 ><<<gHU, 128, 0, stream>>>(xrb, tmWrT, srb, nullptr, nullptr, MM, HU, HH);

    // WKV recurrence (B*HU independent channel scans) -> rwkv bf16
    wkv_scan<<<(BB * HU) / 256, 256, 0, stream>>>(kf, kvf, srb, tm_dec, tm_fst, rwkvb);

    // output projection + residual: x2 = xln0 + rwkv @ Wo
    gemm_bf16_wmma<EPI_RESID, float><<<gH, 128, 0, stream>>>(rwkvb, tmWoT, x2f, xln0, nullptr, MM, HH, HU);

    // channel mix
    ln_kernel<<<MM, 256, 0, stream>>>(x2f, xn, post_g, post_b);
    shift_mix2<<<(MM * HH) / 256, 256, 0, stream>>>(xn, cm_mk, cm_mr, xkb, xrb);
    gemm_bf16_wmma<EPI_SILUB, bf16 ><<<gHU, 128, 0, stream>>>(xkb, cmWkT, kkb, nullptr, nullptr, MM, HU, HH);
    gemm_bf16_wmma<EPI_SIGF,  float><<<gH,  128, 0, stream>>>(xrb, cmWrT, rrf, nullptr, nullptr, MM, HH, HH);
    // out = x2 + sigmoid(xr@cm_Wr) + silu(xk@cm_Wk) @ cm_Wv
    gemm_bf16_wmma<EPI_RESID2, float><<<gH, 128, 0, stream>>>(kkb, cmWvT, out, x2f, rrf, MM, HH, HU);

    (void)in_sizes; (void)n_in; (void)out_size; (void)ws_size;
}
